// DCNv4Block_82205674045878
// MI455X (gfx1250) — matrix operand order repaired and checked by reading, not compiled, not run on hardware
//
#include <hip/hip_runtime.h>
#include <hip/hip_bf16.h>
#include <math.h>

// ---------------- problem constants ----------------
#define NB   8
#define CCH  256
#define HH   64
#define WW   64
#define LL   (HH * WW)          // 4096
#define NTOK (NB * LL)          // 32768
#define GG   8
#define PP   9
#define GC   32
#define NOM  (GG * PP * 3)      // 216
#define NCAT (CCH + NOM)        // 472
#define NPAD 512                // padded GEMM1 output width
#define KK   256                // GEMM reduction dim (both GEMMs)

#define AS1 __attribute__((address_space(1)))
#define AS3 __attribute__((address_space(3)))

typedef __attribute__((ext_vector_type(16))) __bf16 v16bf;
typedef __attribute__((ext_vector_type(8)))  __bf16 v8bf;
typedef __attribute__((ext_vector_type(8)))  float  v8f;
typedef __attribute__((__vector_size__(16))) int    v4i;   // matches builtin param

union Frag16 { v16bf v; v8bf h[2]; };

__device__ __forceinline__ __bf16 f2bf(float f) {
  unsigned u = __builtin_bit_cast(unsigned, f);
  u += 0x7FFFu + ((u >> 16) & 1u);          // round-to-nearest-even
  unsigned short s = (unsigned short)(u >> 16);
  return __builtin_bit_cast(__bf16, s);
}

// ---------------- kernel 1: weight / bias packing ----------------
__global__ __launch_bounds__(256)
void prep_weights(const float* __restrict__ Wv, const float* __restrict__ Wom,
                  const float* __restrict__ Wo, const float* __restrict__ bv,
                  const float* __restrict__ bom,
                  __bf16* __restrict__ WcatT, __bf16* __restrict__ WoT,
                  float* __restrict__ biascat) {
  int idx = blockIdx.x * blockDim.x + threadIdx.x;
  const int NW1 = NPAD * CCH;
  const int NW2 = CCH * CCH;
  if (idx < NW1) {
    int j = idx >> 8, k = idx & 255;
    float v = 0.f;
    if (j < CCH)       v = Wv[k * CCH + j];
    else if (j < NCAT) v = Wom[k * NOM + (j - CCH)];
    WcatT[(size_t)j * CCH + k] = f2bf(v);
  } else if (idx < NW1 + NW2) {
    int t = idx - NW1;
    int j = t >> 8, k = t & 255;
    WoT[(size_t)j * CCH + k] = f2bf(Wo[k * CCH + j]);
  } else if (idx < NW1 + NW2 + NPAD) {
    int j = idx - (NW1 + NW2);
    float v = 0.f;
    if (j < CCH)       v = bv[j];
    else if (j < NCAT) v = bom[j - CCH];
    biascat[j] = v;
  }
}

// ---------------- kernel 2: NCHW f32 -> (n,L,C) bf16 ----------------
__global__ __launch_bounds__(256)
void x_to_seq_bf16(const float* __restrict__ x, __bf16* __restrict__ seqb) {
  __shared__ float tile[32][33];
  const int n  = blockIdx.z;
  const int l0 = blockIdx.x * 32;
  const int c0 = blockIdx.y * 32;
  const int tx = threadIdx.x, ty = threadIdx.y;
#pragma unroll
  for (int i = 0; i < 4; ++i) {
    int c = c0 + ty + i * 8;
    tile[ty + i * 8][tx] = x[((size_t)n * CCH + c) * LL + l0 + tx];
  }
  __syncthreads();
#pragma unroll
  for (int i = 0; i < 4; ++i) {
    int l = l0 + ty + i * 8;
    seqb[((size_t)n * LL + l) * CCH + c0 + tx] = f2bf(tile[tx][ty + i * 8]);
  }
}

// ---------------- kernel 3/5: bf16 WMMA GEMM with async-LDS B staging ----
// C[M x ldc] (f32) = A[M x 256] (bf16 row-major) * Bt[N x 256] (bf16, B^T)
//   + bias[N].
// Block = 256 threads = 8 waves; block tile 128(M) x 64(N).
// The 64x256 B tile (32 KB) is staged once into LDS with
// GLOBAL_LOAD_ASYNC_TO_LDS_B128 (ASYNCcnt) and shared by all 8 waves.
__global__ __launch_bounds__(256)
void gemm_bf16_wmma(const __bf16* __restrict__ A, const __bf16* __restrict__ Bt,
                    const float* __restrict__ bias, float* __restrict__ Cout,
                    int M, int ldc) {
  __shared__ __align__(16) __bf16 bsh[64 * KK];   // 32 KB

  const int lane = threadIdx.x & 31;
  const int wid  = threadIdx.x >> 5;     // 0..7
  const int l16  = lane & 15;
  const int half = lane >> 4;            // 0|1
  const int m0 = blockIdx.x * 128 + wid * 16;
  const int n0 = blockIdx.y * 64;

  // ---- async stage of the block's B tile: 2048 x 16B chunks, 8 per thread
  {
    const __bf16* btile = Bt + (size_t)n0 * KK;
#pragma unroll
    for (int i = 0; i < 8; ++i) {
      const int ch = threadIdx.x + i * 256;     // chunk id 0..2047
      const int j  = ch >> 5;                   // B row (out column)
      const int e  = (ch & 31) * 8;             // element within row
      __builtin_amdgcn_global_load_async_to_lds_b128(
          (AS1 v4i*)(btile + (size_t)j * KK + e),
          (AS3 v4i*)((char*)bsh + (size_t)ch * 16), 0, 0);
    }
    __builtin_amdgcn_s_wait_asynccnt(0);
    __syncthreads();
  }

  v8f acc[4] = {v8f{0}, v8f{0}, v8f{0}, v8f{0}};

  // ISA 16-bit A layout: lanes 0-15 hold K[0..7]+K[16..23],
  //                      lanes 16-31 hold K[8..15]+K[24..31] of row m0+l16.
  const __bf16* arow = A + (size_t)(m0 + l16) * KK;

  for (int k0 = 0; k0 < KK; k0 += 32) {
    __builtin_prefetch(arow + k0 + 64, 0, 3);   // near-cache prefetch on A
    Frag16 a;
    a.h[0] = *(const v8bf*)(arow + k0 + half * 8);
    a.h[1] = *(const v8bf*)(arow + k0 + 16 + half * 8);
#pragma unroll
    for (int j = 0; j < 4; ++j) {
      // B fragment from LDS: lane holds column j*16+l16 of the tile,
      // K values matching A (lanes 0-15: K[0..7]+K[16..23], 16-31: K[8..15]+K[24..31]).
      const __bf16* brow = bsh + (size_t)(j * 16 + l16) * KK + k0 + half * 8;
      Frag16 b;
      b.h[0] = *(const v8bf*)(brow);
      b.h[1] = *(const v8bf*)(brow + 16);
      acc[j] = __builtin_amdgcn_wmma_f32_16x16x32_bf16(
          false, a.v, false, b.v, (short)0, acc[j], false, false);
    }
  }

#pragma unroll
  for (int j = 0; j < 4; ++j) {
    const int col = n0 + j * 16 + l16;
    const float bj = bias[col];
    float* crow = Cout + (size_t)(m0 + half * 8) * ldc + col;
#pragma unroll
    for (int r = 0; r < 8; ++r) {         // D: vgpr r -> row m0 + half*8 + r
      crow[(size_t)r * ldc] = acc[j][r] + bj;
    }
  }
}

// ---------------- kernel 4: DCNv4 bilinear sampling ----------------
// valom row (per token, ld=NPAD): [0,256) = val (G,GC), [256,472) = (G,P,3).
// One block per token; wave g handles group g; lane = channel within group.
__global__ __launch_bounds__(256)
void dcn_sample(const float* __restrict__ valom, __bf16* __restrict__ sampb) {
  const int token = blockIdx.x;           // n*L + l
  const int g    = threadIdx.x >> 5;
  const int lane = threadIdx.x & 31;
  const int n = token >> 12;              // / 4096
  const int l = token & (LL - 1);
  const int h = l >> 6, w = l & 63;

  const float* om = valom + (size_t)token * NPAD + CCH + g * (PP * 3);
  const float* vb = valom + (size_t)n * LL * NPAD + g * GC + lane;

  float acc = 0.f;
#pragma unroll
  for (int p = 0; p < PP; ++p) {
    const float offx = om[p * 3 + 0];
    const float offy = om[p * 3 + 1];
    const float msk  = om[p * 3 + 2];
    const float px = (float)(w + (p % 3) - 1) + offx;
    const float py = (float)(h + (p / 3) - 1) + offy;
    const float x0 = floorf(px), y0 = floorf(py);
    const float wx = px - x0, wy = py - y0;
    const int ix = (int)x0, iy = (int)y0;

    float v00 = 0.f, v10 = 0.f, v01 = 0.f, v11 = 0.f;
    const bool xin0 = (ix >= 0) & (ix < WW);
    const bool xin1 = (ix + 1 >= 0) & (ix + 1 < WW);
    const bool yin0 = (iy >= 0) & (iy < HH);
    const bool yin1 = (iy + 1 >= 0) & (iy + 1 < HH);
    if (xin0 & yin0) v00 = vb[(size_t)(iy * WW + ix) * NPAD];
    if (xin1 & yin0) v10 = vb[(size_t)(iy * WW + ix + 1) * NPAD];
    if (xin0 & yin1) v01 = vb[(size_t)((iy + 1) * WW + ix) * NPAD];
    if (xin1 & yin1) v11 = vb[(size_t)((iy + 1) * WW + ix + 1) * NPAD];

    const float bil = (1.f - wx) * (1.f - wy) * v00 + wx * (1.f - wy) * v10 +
                      (1.f - wx) * wy * v01 + wx * wy * v11;
    acc = fmaf(msk, bil, acc);
  }
  sampb[(size_t)token * CCH + g * GC + lane] = f2bf(acc);
}

// ---------------- kernel 6: epilogue (transpose + BN + GELU + residual) ----
__global__ __launch_bounds__(256)
void epilogue_nchw(const float* __restrict__ outseq, const float* __restrict__ x,
                   const float* __restrict__ gamma, const float* __restrict__ beta,
                   float* __restrict__ out) {
  __shared__ float tile[32][33];
  const int n  = blockIdx.z;
  const int l0 = blockIdx.x * 32;
  const int c0 = blockIdx.y * 32;
  const int tx = threadIdx.x, ty = threadIdx.y;
#pragma unroll
  for (int i = 0; i < 4; ++i) {
    int l = l0 + ty + i * 8;
    tile[ty + i * 8][tx] = outseq[((size_t)n * LL + l) * CCH + c0 + tx];
  }
  __syncthreads();
  const float inv = 0.99999500003749969f;   // 1/sqrt(1 + 1e-5)
#pragma unroll
  for (int i = 0; i < 4; ++i) {
    const int c = c0 + ty + i * 8;
    const int l = l0 + tx;
    const float feat = tile[tx][ty + i * 8];
    const float z  = fmaf(feat, gamma[c] * inv, beta[c]);
    const float ge = 0.5f * z * (1.f + erff(z * 0.70710678118654752f));
    const size_t idx = ((size_t)n * CCH + c) * LL + l;
    out[idx] = ge + x[idx];
  }
}

// ---------------- host launcher ----------------
extern "C" void kernel_launch(void* const* d_in, const int* in_sizes, int n_in,
                              void* d_out, int out_size, void* d_ws, size_t ws_size,
                              hipStream_t stream) {
  const float* x     = (const float*)d_in[0];
  const float* Wv    = (const float*)d_in[1];
  const float* bv    = (const float*)d_in[2];
  const float* Wom   = (const float*)d_in[3];
  const float* bom   = (const float*)d_in[4];
  const float* Wo    = (const float*)d_in[5];
  const float* bo    = (const float*)d_in[6];
  const float* gam   = (const float*)d_in[7];
  const float* bet   = (const float*)d_in[8];
  float* out = (float*)d_out;

  // workspace carve-out (256B aligned)
  char* ws = (char*)d_ws;
  size_t off = 0;
  auto carve = [&](size_t bytes) {
    void* p = ws + off;
    off = (off + bytes + 255) & ~(size_t)255;
    return p;
  };
  __bf16* seqb    = (__bf16*)carve((size_t)NTOK * CCH * 2);   // 16.8 MB
  __bf16* WcatT   = (__bf16*)carve((size_t)NPAD * CCH * 2);   // 256 KB
  __bf16* WoT     = (__bf16*)carve((size_t)CCH * CCH * 2);    // 128 KB
  float*  biascat = (float*) carve((size_t)NPAD * 4);
  float*  valom   = (float*) carve((size_t)NTOK * NPAD * 4);  // 67 MB
  __bf16* sampb   = (__bf16*)carve((size_t)NTOK * CCH * 2);   // 16.8 MB
  float*  outseq  = (float*) carve((size_t)NTOK * CCH * 4);   // 33.5 MB

  // 1) pack weights
  {
    const int total = NPAD * CCH + CCH * CCH + NPAD;
    prep_weights<<<(total + 255) / 256, 256, 0, stream>>>(
        Wv, Wom, Wo, bv, bom, WcatT, WoT, biascat);
  }
  // 2) x -> seq bf16
  x_to_seq_bf16<<<dim3(LL / 32, CCH / 32, NB), dim3(32, 8), 0, stream>>>(x, seqb);
  // 3) GEMM1: val + om (N = 512 padded)
  gemm_bf16_wmma<<<dim3(NTOK / 128, NPAD / 64), 256, 0, stream>>>(
      seqb, WcatT, biascat, valom, NTOK, NPAD);
  // 4) deformable sampling -> bf16
  dcn_sample<<<dim3(NTOK), 256, 0, stream>>>(valom, sampb);
  // 5) GEMM2: output projection
  gemm_bf16_wmma<<<dim3(NTOK / 128, CCH / 64), 256, 0, stream>>>(
      sampb, WoT, bo, outseq, NTOK, CCH);
  // 6) transpose + BN + GELU + residual
  epilogue_nchw<<<dim3(LL / 32, CCH / 32, NB), dim3(32, 8), 0, stream>>>(
      outseq, x, gam, bet, out);
}
